// CausalSelfAttention_25958782337675
// MI455X (gfx1250) — matrix operand order repaired and checked by reading, not compiled, not run on hardware
//
#include <hip/hip_runtime.h>

// ---------------------------------------------------------------------------
// Causal self-attention for MI455X (gfx1250), bf16 WMMA pipeline, software-
// pipelined fragment loads.   B=2, T=2048, C=1024, H=16, D=64
// ---------------------------------------------------------------------------

typedef __attribute__((ext_vector_type(16))) __bf16 v16bf;
typedef __attribute__((ext_vector_type(8)))  float  v8f;

#define T_SEQ 2048
#define CDIM  1024
#define HEADS 16
#define DHEAD 64
#define BATCH 2

#define LOG2E 1.4426950408889634f

union BFrag {
    v16bf bf;
    uint4 q[2];
};

__device__ __forceinline__ unsigned short f2bf(float x) {
    unsigned int u = __float_as_uint(x);
    u += 0x7FFFu + ((u >> 16) & 1u);      // round-to-nearest-even
    return (unsigned short)(u >> 16);
}

__device__ __forceinline__ v8f wmma_bf16(v16bf a, v16bf b, v8f c) {
    return __builtin_amdgcn_wmma_f32_16x16x32_bf16(
        /*neg_a=*/false, a, /*neg_b=*/false, b,
        /*c_mod=*/(short)0, c, /*reuse_a=*/false, /*reuse_b=*/false);
}

// Fragment gather at a precomputed per-lane pointer; second half at a
// compile-time element offset (16 for both A- and B-frags).
template <int OFF1>
__device__ __forceinline__ v16bf frag_at(const unsigned short* p) {
    BFrag f;
    f.q[0] = *(const uint4*)(p);
    f.q[1] = *(const uint4*)(p + OFF1);
    return f.bf;
}

// A fragment: 16(M) x 32(K) bf16 from row-major src[m][k], leading dim ld.
__device__ __forceinline__ v16bf load_frag_a(const unsigned short* __restrict__ base,
                                             int ld, int m0, int k0) {
    int lane = threadIdx.x & 31;
    int m  = m0 + (lane & 15);
    int kh = (lane >> 4) << 3;            // 0 or 8
    return frag_at<16>(base + (size_t)m * ld + k0 + kh);
}

// B fragment: 32(K) x 16(N) bf16 where B[k][n] = src[n][k] (row-major over k).
__device__ __forceinline__ v16bf load_frag_b(const unsigned short* __restrict__ base,
                                             int ld, int n0, int k0) {
    int lane = threadIdx.x & 31;
    int n  = n0 + (lane & 15);
    int ks = (lane >> 4) << 3;            // 0 or 8
    return frag_at<16>(base + (size_t)n * ld + k0 + ks);
}

// ---------------------------------------------------------------------------
// Kernel 1: fp32 -> bf16 conversion, 4-wide (b128 in, b64 out)
// ---------------------------------------------------------------------------
__global__ void cvt_bf16_kernel(const float* __restrict__ src,
                                unsigned int* __restrict__ dst, int n4) {
    int i = blockIdx.x * blockDim.x + threadIdx.x;
    int stride = gridDim.x * blockDim.x;
    for (; i < n4; i += stride) {
        float4 f = *(const float4*)(src + (size_t)i * 4);
        unsigned int lo = (unsigned int)f2bf(f.x) | ((unsigned int)f2bf(f.y) << 16);
        unsigned int hi = (unsigned int)f2bf(f.z) | ((unsigned int)f2bf(f.w) << 16);
        dst[(size_t)i * 2]     = lo;
        dst[(size_t)i * 2 + 1] = hi;
    }
}

// ---------------------------------------------------------------------------
// Kernel 2: Y = X @ W^T + bias, bf16 out in head-split layout.
// One wave computes a 32(M) x 32(N) tile: 4 WMMAs per k-step of 32,
// double-buffered loads off TWO per-lane base pointers (all 8 b128 loads at
// immediate offsets; bases advance by 64B per k-step -> no address spills).
//   transpose_out==0:  out[((b*H+h)*T + t)*D + d]      (Q, K)
//   transpose_out==1:  out[((b*H+h)*D + d)*T + t]      (V transposed)
// ---------------------------------------------------------------------------
__global__ void qkv_gemm_kernel(const unsigned short* __restrict__ Xb,
                                const unsigned short* __restrict__ Wb,
                                const float* __restrict__ bias,
                                unsigned short* __restrict__ outp,
                                int transpose_out) {
    int wave = blockIdx.x * (blockDim.x >> 5) + (threadIdx.x >> 5);
    int mt = wave >> 5;                   // 0..127  (M tiles of 32)
    int ns = wave & 31;                   // 0..31   (N tiles of 32)
    int m0 = mt << 5;
    int n0 = ns << 5;

    int lane = threadIdx.x & 31;
    int nl = lane & 15;
    int hi = lane >> 4;                   // half-wave select

    // Per-lane base pointers; every load below is base + immediate.
    const unsigned short* pa = Xb + (size_t)(m0 + nl) * CDIM + (hi << 3);
    const unsigned short* pb = Wb + (size_t)(n0 + nl) * CDIM + (hi << 3);
    const int ROW16 = 16 * CDIM;          // 32768 B: second M/N tile offset

    v8f acc[2][2] = {};

    // prologue: fragments for k = 0
    v16bf a0 = frag_at<16>(pa);
    v16bf a1 = frag_at<16>(pa + ROW16);
    v16bf b0 = frag_at<16>(pb);
    v16bf b1 = frag_at<16>(pb + ROW16);

    // steady state: issue k-step loads, then consume previous step's frags
    for (int k = 32; k < CDIM; k += 32) {
        pa += 32; pb += 32;
        v16bf a0n = frag_at<16>(pa);
        v16bf a1n = frag_at<16>(pa + ROW16);
        v16bf b0n = frag_at<16>(pb);
        v16bf b1n = frag_at<16>(pb + ROW16);

        acc[0][0] = wmma_bf16(a0, b0, acc[0][0]);
        acc[0][1] = wmma_bf16(a0, b1, acc[0][1]);
        acc[1][0] = wmma_bf16(a1, b0, acc[1][0]);
        acc[1][1] = wmma_bf16(a1, b1, acc[1][1]);

        a0 = a0n; a1 = a1n; b0 = b0n; b1 = b1n;
    }
    // epilogue k-step
    acc[0][0] = wmma_bf16(a0, b0, acc[0][0]);
    acc[0][1] = wmma_bf16(a0, b1, acc[0][1]);
    acc[1][0] = wmma_bf16(a1, b0, acc[1][0]);
    acc[1][1] = wmma_bf16(a1, b1, acc[1][1]);

    int mh = hi << 3;                     // row offset 0 or 8
#pragma unroll
    for (int u = 0; u < 2; ++u) {
        int m0u = m0 + (u << 4);
#pragma unroll
        for (int t = 0; t < 2; ++t) {
            int n = n0 + (t << 4) + nl;
            float bv = bias[n];
            int h = n >> 6, d = n & (DHEAD - 1);
#pragma unroll
            for (int v = 0; v < 8; ++v) {
                int m = m0u + v + mh;
                int bb_ = m >> 11, tt = m & (T_SEQ - 1);
                float val = acc[u][t][v] + bv;
                size_t idx = transpose_out
                    ? ((size_t)((bb_ * HEADS + h) * DHEAD + d) * T_SEQ + tt)
                    : ((size_t)((bb_ * HEADS + h) * T_SEQ + tt) * DHEAD + d);
                outp[idx] = f2bf(val);
            }
        }
    }
}

// ---------------------------------------------------------------------------
// Kernel 3: flash-style causal attention, one wave per 16-query tile.
// Pipelining: S-WMMAs -> issue next K frags + current V frags + prefetch ->
// softmax VALU stretch (hides load latency) -> P LDS relayout -> PV WMMAs.
// ---------------------------------------------------------------------------
__global__ void attn_kernel(const unsigned short* __restrict__ Qb,
                            const unsigned short* __restrict__ Kb,
                            const unsigned short* __restrict__ Vt,
                            const float* __restrict__ amask,
                            float* __restrict__ outp) {
    __shared__ unsigned short pbuf[4][16 * 32];

    int wid  = threadIdx.x >> 5;
    int wave = blockIdx.x * (blockDim.x >> 5) + wid;
    int qt = wave & 127;                  // query tile within head
    int bh = wave >> 7;                   // 0..31
    int b  = bh >> 4;
    int h  = bh & 15;
    int q0 = qt << 4;

    const unsigned short* Qh = Qb + (size_t)bh * T_SEQ * DHEAD;
    const unsigned short* Kh = Kb + (size_t)bh * T_SEQ * DHEAD;
    const unsigned short* Vh = Vt + (size_t)bh * DHEAD * T_SEQ;

    int lane = threadIdx.x & 31;
    int nl = lane & 15;
    int mh = (lane >> 4) << 3;            // 0 or 8 (row half / K half)

    // Q fragments (K-dim 64 = two 16x32 A-frags), resident for whole loop.
    v16bf qa0 = load_frag_a(Qh, DHEAD, q0, 0);
    v16bf qa1 = load_frag_a(Qh, DHEAD, q0, 32);

    v8f o[4] = {};
    float mrun[8], lrun[8];
#pragma unroll
    for (int v = 0; v < 8; ++v) { mrun[v] = -3.0e38f; lrun[v] = 0.0f; }

    unsigned short* lds = pbuf[wid];

    int nblocks = (q0 + 16 + 31) >> 5;    // causal limit, 32 keys per block
    int jlast   = (nblocks - 1) << 5;

    // prologue: K fragments for block 0
    v16bf kb00 = load_frag_b(Kh, DHEAD, 0,  0);
    v16bf kb01 = load_frag_b(Kh, DHEAD, 0,  32);
    v16bf kb10 = load_frag_b(Kh, DHEAD, 16, 0);
    v16bf kb11 = load_frag_b(Kh, DHEAD, 16, 32);

    for (int kt = 0; kt < nblocks; ++kt) {
        int j0 = kt << 5;

        // ---- S = Q K^T on current K fragments ----
        v8f s[2] = {};
        s[0] = wmma_bf16(qa0, kb00, s[0]);
        s[0] = wmma_bf16(qa1, kb01, s[0]);
        s[1] = wmma_bf16(qa0, kb10, s[1]);
        s[1] = wmma_bf16(qa1, kb11, s[1]);

        // ---- issue next-block K loads + this-block V loads now ----
        int jn = (j0 + 32 < T_SEQ) ? j0 + 32 : jlast;   // clamped (dup on last)
        kb00 = load_frag_b(Kh, DHEAD, jn,      0);
        kb01 = load_frag_b(Kh, DHEAD, jn,      32);
        kb10 = load_frag_b(Kh, DHEAD, jn + 16, 0);
        kb11 = load_frag_b(Kh, DHEAD, jn + 16, 32);
        v16bf vb[4];
#pragma unroll
        for (int dt = 0; dt < 4; ++dt)
            vb[dt] = load_frag_b(Vh, T_SEQ, dt << 4, j0);
        if (j0 + 64 < T_SEQ)   // prefetch K rows two blocks ahead (one row/lane)
            __builtin_prefetch(Kh + (size_t)(j0 + 64 + lane) * DHEAD, 0, 3);

        // ---- scale*log2e, additive padding mask, causal mask ----
#pragma unroll
        for (int t = 0; t < 2; ++t) {
            int gk = j0 + (t << 4) + nl;
            float mk = amask[b * T_SEQ + gk] * LOG2E;
#pragma unroll
            for (int v = 0; v < 8; ++v) {
                int gq = q0 + v + mh;
                float val = s[t][v] * (0.125f * LOG2E) + mk;
                s[t][v] = (gk > gq) ? -3.0e38f : val;
            }
        }

        // ---- per-row max across the 16 lanes of each half-wave ----
        float rmax[8];
#pragma unroll
        for (int v = 0; v < 8; ++v) rmax[v] = fmaxf(s[0][v], s[1][v]);
#pragma unroll
        for (int off = 1; off < 16; off <<= 1)
#pragma unroll
            for (int v = 0; v < 8; ++v)
                rmax[v] = fmaxf(rmax[v], __shfl_xor(rmax[v], off, 32));

        float alpha[8];
#pragma unroll
        for (int v = 0; v < 8; ++v) {
            float mnew = fmaxf(mrun[v], rmax[v]);
            alpha[v] = exp2f(mrun[v] - mnew);
            mrun[v] = mnew;
        }

        // ---- P = 2^(S - m), row sums, stage P (bf16) to LDS ----
        float rsum[8];
#pragma unroll
        for (int v = 0; v < 8; ++v) rsum[v] = 0.0f;
#pragma unroll
        for (int t = 0; t < 2; ++t)
#pragma unroll
            for (int v = 0; v < 8; ++v) {
                float p = exp2f(s[t][v] - mrun[v]);
                rsum[v] += p;
                lds[(v + mh) * 32 + (t << 4) + nl] = f2bf(p);
            }
#pragma unroll
        for (int off = 1; off < 16; off <<= 1)
#pragma unroll
            for (int v = 0; v < 8; ++v)
                rsum[v] += __shfl_xor(rsum[v], off, 32);
#pragma unroll
        for (int v = 0; v < 8; ++v) lrun[v] = lrun[v] * alpha[v] + rsum[v];

        // ---- rescale running O ----
#pragma unroll
        for (int dt = 0; dt < 4; ++dt)
#pragma unroll
            for (int v = 0; v < 8; ++v)
                o[dt][v] = o[dt][v] * alpha[v];

        // ---- reload P in A-fragment layout (wave-private LDS, DS in-order) ----
        BFrag pf;
        {
            const unsigned short* pp = lds + nl * 32 + mh; // row=nl, khalf=mh
            pf.q[0] = *(const uint4*)(pp);
            pf.q[1] = *(const uint4*)(pp + 16);
        }

        // ---- O += P * V   (V transposed: Vh[d*T + j], contiguous in j) ----
#pragma unroll
        for (int dt = 0; dt < 4; ++dt)
            o[dt] = wmma_bf16(pf.bf, vb[dt], o[dt]);
    }

    // ---- normalize and store fp32 output [B,T,C] ----
    float invl[8];
#pragma unroll
    for (int v = 0; v < 8; ++v) invl[v] = 1.0f / lrun[v];
#pragma unroll
    for (int dt = 0; dt < 4; ++dt) {
#pragma unroll
        for (int v = 0; v < 8; ++v) {
            int t = q0 + v + mh;
            int c = (h << 6) + (dt << 4) + nl;
            outp[(size_t)(b * T_SEQ + t) * CDIM + c] = o[dt][v] * invl[v];
        }
    }
}

// ---------------------------------------------------------------------------
// Host launcher
// ---------------------------------------------------------------------------
extern "C" void kernel_launch(void* const* d_in, const int* in_sizes, int n_in,
                              void* d_out, int out_size, void* d_ws, size_t ws_size,
                              hipStream_t stream) {
    const float* X     = (const float*)d_in[0];
    const float* amask = (const float*)d_in[1];
    const float* Wq    = (const float*)d_in[2];
    const float* bq    = (const float*)d_in[3];
    const float* Wk    = (const float*)d_in[4];
    const float* bk    = (const float*)d_in[5];
    const float* Wv    = (const float*)d_in[6];
    const float* bv    = (const float*)d_in[7];
    float* out = (float*)d_out;

    char* ws = (char*)d_ws;
    unsigned short* Xb  = (unsigned short*)(ws);                        //  8 MB
    unsigned short* Wqb = (unsigned short*)(ws + ((size_t)8  << 20));   //  2 MB
    unsigned short* Wkb = (unsigned short*)(ws + ((size_t)10 << 20));   //  2 MB
    unsigned short* Wvb = (unsigned short*)(ws + ((size_t)12 << 20));   //  2 MB
    unsigned short* Qb  = (unsigned short*)(ws + ((size_t)14 << 20));   //  8 MB
    unsigned short* Kb  = (unsigned short*)(ws + ((size_t)22 << 20));   //  8 MB
    unsigned short* Vt  = (unsigned short*)(ws + ((size_t)30 << 20));   //  8 MB

    const int nx4 = (BATCH * T_SEQ * CDIM) / 4;  // 1,048,576 quads
    const int nw4 = (CDIM * CDIM) / 4;           //   262,144 quads
    cvt_bf16_kernel<<<1024, 256, 0, stream>>>(X,  (unsigned int*)Xb,  nx4);
    cvt_bf16_kernel<<<1024, 256, 0, stream>>>(Wq, (unsigned int*)Wqb, nw4);
    cvt_bf16_kernel<<<1024, 256, 0, stream>>>(Wk, (unsigned int*)Wkb, nw4);
    cvt_bf16_kernel<<<1024, 256, 0, stream>>>(Wv, (unsigned int*)Wvb, nw4);

    // 4096 (32x32) tiles per GEMM; 4 waves per 128-thread block.
    qkv_gemm_kernel<<<1024, 128, 0, stream>>>(Xb, Wqb, bq, Qb, 0);
    qkv_gemm_kernel<<<1024, 128, 0, stream>>>(Xb, Wkb, bk, Kb, 0);
    qkv_gemm_kernel<<<1024, 128, 0, stream>>>(Xb, Wvb, bv, Vt, 1);

    // 4096 query-tile waves (2*16*128); 4 waves per block.
    attn_kernel<<<1024, 128, 0, stream>>>(Qb, Kb, Vt, amask, out);
}
